// G2GDecoder_76459007804090
// MI455X (gfx1250) — matrix operand order repaired and checked by reading, not compiled, not run on hardware
//
#include <hip/hip_runtime.h>

// CDNA5 / gfx1250 implementation. wave32, WMMA f32_16x16x32_f16 with fp32 acc.
#define D 128

typedef __attribute__((ext_vector_type(16))) _Float16 v16h;
typedef __attribute__((ext_vector_type(8)))  _Float16 v8h;
typedef __attribute__((ext_vector_type(8)))  float    v8f;
typedef __attribute__((ext_vector_type(4)))  float    v4f;

// ---------------------------------------------------------------------------
// Generic fused WMMA GEMM: out = epi( A0@W0 + A1@W1 + A2@W2 + bias )
// Block = 256 threads (8 waves). Each wave computes TWO 16-row x 128-col
// stripes (block covers 256 rows), so every B fragment feeds two WMMAs and
// the 32KB LDS weight stage amortizes over 2x the rows. B fragments are
// double-buffered in registers so ds_load of tile nt+1 overlaps WMMA of nt.
//
// epi 0: out[r*ldOut+c] = act(v)                (act: 0 none, 1 sigmoid, 2 relu)
// epi 1: r = sigmoid(v); atomicAdd(srh[lgd[row]][c], r * emsg[lgs[row]][c])
// epi 2: h~ = tanh(v); out[row][c] = (1-zb)*sb + zb*h~   (out may alias sb)
// ---------------------------------------------------------------------------
__global__ __launch_bounds__(256) void wmma_gemm(
    int nRows, int nColsTotal, int ldW, int ldOut,
    const float* __restrict__ A0, const int* __restrict__ g0, const float* __restrict__ W0,
    const float* __restrict__ A1, const int* __restrict__ g1, const float* __restrict__ W1,
    const float* __restrict__ A2, const int* __restrict__ g2, const float* __restrict__ W2,
    const float* __restrict__ bias, int act, int epi,
    float* __restrict__ out,
    const float* __restrict__ emsg, const int* __restrict__ lgs, const int* __restrict__ lgd,
    float* __restrict__ srh,
    const float* __restrict__ zb, const float* __restrict__ sb)
{
  __shared__ _Float16 Wlds[D * D];   // 32 KB: weight chunk, transposed [n][k]
  const int tid  = threadIdx.x;
  const int wave = tid >> 5;
  const int lane = tid & 31;
  const int h    = lane >> 4;        // half-wave select
  const int ln   = lane & 15;
  const int row0 = blockIdx.x * 256 + wave * 32;   // two stripes: row0, row0+16
  const int col0 = blockIdx.y * 128;
  int ncols = nColsTotal - col0; if (ncols > 128) ncols = 128;

  v8f acc[2][8];
#pragma unroll
  for (int st = 0; st < 2; ++st)
#pragma unroll
    for (int i = 0; i < 8; ++i)
#pragma unroll
      for (int jj = 0; jj < 8; ++jj) acc[st][i][jj] = 0.f;

  for (int j = 0; j < 3; ++j) {
    const float* A = (j == 0) ? A0 : (j == 1) ? A1 : A2;
    if (A == nullptr) break;
    const int*   g = (j == 0) ? g0 : (j == 1) ? g1 : g2;
    const float* W = (j == 0) ? W0 : (j == 1) ? W1 : W2;

    __syncthreads();
    for (int i = tid; i < D * D; i += 256) {
      int n = i >> 7, k = i & 127;
      float wv = (n < ncols) ? W[(size_t)k * ldW + col0 + n] : 0.f;
      Wlds[n * D + k] = (_Float16)wv;     // transposed store
    }
    __syncthreads();

    // per-stripe gathered A row pointers (two half-waves read the same rows)
    int r0 = row0 + ln;
    int r1 = row0 + 16 + ln;
    int rc0 = (r0 < nRows) ? r0 : 0;
    int rc1 = (r1 < nRows) ? r1 : 0;
    int ri0 = g ? g[rc0] : rc0;
    int ri1 = g ? g[rc1] : rc1;
    const float* arow0 = A + (size_t)ri0 * D;
    const float* arow1 = A + (size_t)ri1 * D;

#pragma unroll
    for (int kk = 0; kk < 4; ++kk) {
      // A fragments (16-bit A 16x32 layout): elems 0..7 -> K=kk*32+8h+i,
      // elems 8..15 -> K=kk*32+16+8h+i
      const int kb = kk * 32 + 8 * h;
      v4f a00 = *(const v4f*)(arow0 + kb);
      v4f a01 = *(const v4f*)(arow0 + kb + 4);
      v4f a02 = *(const v4f*)(arow0 + kb + 16);
      v4f a03 = *(const v4f*)(arow0 + kb + 20);
      v4f a10 = *(const v4f*)(arow1 + kb);
      v4f a11 = *(const v4f*)(arow1 + kb + 4);
      v4f a12 = *(const v4f*)(arow1 + kb + 16);
      v4f a13 = *(const v4f*)(arow1 + kb + 20);
      v16h af0, af1;
#pragma unroll
      for (int i = 0; i < 4; ++i) {
        af0[i]      = (_Float16)a00[i];
        af0[4 + i]  = (_Float16)a01[i];
        af0[8 + i]  = (_Float16)a02[i];
        af0[12 + i] = (_Float16)a03[i];
        af1[i]      = (_Float16)a10[i];
        af1[4 + i]  = (_Float16)a11[i];
        af1[8 + i]  = (_Float16)a12[i];
        af1[12 + i] = (_Float16)a13[i];
      }
      // B fragment double buffer: elem i<8 = W[K = kk*32 + 8h + i][n],
      // elem 8+i = W[K = kk*32 + 16 + 8h + i][n], from Wlds[n][k]
      const int bofs = kk * 32 + 8 * h;
      v8h bc0 = *(const v8h*)&Wlds[(0 * 16 + ln) * D + bofs];
      v8h bc1 = *(const v8h*)&Wlds[(0 * 16 + ln) * D + bofs + 16];
#pragma unroll
      for (int nt = 0; nt < 8; ++nt) {
        v8h bn0 = bc0, bn1 = bc1;
        if (nt < 7) {  // issue next tile's LDS loads before this tile's WMMAs
          bn0 = *(const v8h*)&Wlds[((nt + 1) * 16 + ln) * D + bofs];
          bn1 = *(const v8h*)&Wlds[((nt + 1) * 16 + ln) * D + bofs + 16];
        }
        v16h bf;
#pragma unroll
        for (int i = 0; i < 8; ++i) { bf[i] = bc0[i]; bf[8 + i] = bc1[i]; }
        acc[0][nt] = __builtin_amdgcn_wmma_f32_16x16x32_f16(
            false, af0, false, bf, (short)0, acc[0][nt], false, false);
        acc[1][nt] = __builtin_amdgcn_wmma_f32_16x16x32_f16(
            false, af1, false, bf, (short)0, acc[1][nt], false, false);
        bc0 = bn0; bc1 = bn1;
      }
    }
  }

  // ---- epilogue: acc VGPR rg -> M = rg + 8h, N = ln ----
#pragma unroll
  for (int st = 0; st < 2; ++st) {
#pragma unroll
    for (int rg = 0; rg < 8; ++rg) {
      int rr = row0 + st * 16 + rg + 8 * h;
      if (rr >= nRows) continue;
      int isrc = 0, idst = 0;
      if (epi == 1) { isrc = lgs[rr]; idst = lgd[rr]; }
#pragma unroll
      for (int nt = 0; nt < 8; ++nt) {
        int cc = nt * 16 + ln;
        if (cc >= ncols) continue;
        int gc = col0 + cc;
        float v = acc[st][nt][rg] + (bias ? bias[gc] : 0.f);
        if (epi == 0) {
          if (act == 1)      v = 1.f / (1.f + __expf(-v));
          else if (act == 2) v = fmaxf(v, 0.f);
          out[(size_t)rr * ldOut + gc] = v;
        } else if (epi == 1) {
          float rv = 1.f / (1.f + __expf(-v));
          float mv = emsg[(size_t)isrc * D + gc];
          atomicAdd(&srh[(size_t)idst * D + gc], rv * mv);
        } else {
          float ht = tanhf(v);
          float zv = zb[(size_t)rr * D + gc];
          float sv = sb[(size_t)rr * D + gc];
          out[(size_t)rr * D + gc] = (1.f - zv) * sv + zv * ht;
        }
      }
    }
  }
}

// ---------------------------------------------------------------------------
// dst[gdst[e]] += src[gsrc[e]]  (rows of 128 floats; 4 floats per thread)
// ---------------------------------------------------------------------------
__global__ __launch_bounds__(256) void scatter_sum(
    const float* __restrict__ src, const int* __restrict__ gsrc,
    const int* __restrict__ gdst, float* __restrict__ dst, int nEdges)
{
  int i = blockIdx.x * 256 + threadIdx.x;
  int e = i >> 5;
  if (e >= nEdges) return;
  int c4 = (i & 31) * 4;
  int s = gsrc[e], dd = gdst[e];
  v4f v = *(const v4f*)(src + (size_t)s * D + c4);
  float* p = dst + (size_t)dd * D + c4;
  atomicAdd(p + 0, v[0]); atomicAdd(p + 1, v[1]);
  atomicAdd(p + 2, v[2]); atomicAdd(p + 3, v[3]);
}

// ---------------------------------------------------------------------------
// Exclusive prefix sum of bnn[0..B) -> offs[0..B], single block of 256.
// ---------------------------------------------------------------------------
__global__ __launch_bounds__(256) void scan_offsets(
    const int* __restrict__ bnn, int* __restrict__ offs, int B)
{
  __shared__ int csum[257];
  int tid = threadIdx.x;
  int chunk = (B + 255) / 256;
  int base = tid * chunk;
  int s = 0;
  for (int i = 0; i < chunk; ++i) { int idx = base + i; if (idx < B) s += bnn[idx]; }
  csum[tid + 1] = s;
  if (tid == 0) csum[0] = 0;
  __syncthreads();
  if (tid == 0) for (int i = 1; i <= 256; ++i) csum[i] += csum[i - 1];
  __syncthreads();
  int run = csum[tid];
  for (int i = 0; i < chunk; ++i) {
    int idx = base + i;
    if (idx < B) { offs[idx] = run; run += bnn[idx]; }
  }
  if (tid == 255) offs[B] = run;
}

// ---------------------------------------------------------------------------
// Dual segment-softmax attention: for graph b with rows [offs[b],offs[b+1]),
//   e_i = dot(x_i, hp_b); p = softmax(e); out_b = sum_i p_i * x_i
// computed simultaneously for (hp1,out1) and (hp2,out2) sharing x traffic.
// One block (128 threads) per graph; n <= 128.
// ---------------------------------------------------------------------------
__global__ __launch_bounds__(128) void attn_dual(
    const float* __restrict__ x, const int* __restrict__ offs,
    const float* __restrict__ hp1, const float* __restrict__ hp2,
    float* __restrict__ out1, float* __restrict__ out2)
{
  __shared__ float sh1[D], sh2[D], p1[D], p2[D], stats[4];
  int b = blockIdx.x, t = threadIdx.x;
  int s0 = offs[b];
  int n = offs[b + 1] - s0;
  if (n > 128) n = 128;
  sh1[t] = hp1[(size_t)b * D + t];
  sh2[t] = hp2[(size_t)b * D + t];
  __syncthreads();
  if (t < n) {
    const float* xr = x + (size_t)(s0 + t) * D;
    float e1 = 0.f, e2 = 0.f;
    for (int k = 0; k < D; ++k) { float xv = xr[k]; e1 += xv * sh1[k]; e2 += xv * sh2[k]; }
    p1[t] = e1; p2[t] = e2;
  }
  __syncthreads();
  if (t == 0) {
    float m = -3.4e38f; for (int i = 0; i < n; ++i) m = fmaxf(m, p1[i]);
    float z = 0.f;      for (int i = 0; i < n; ++i) z += __expf(p1[i] - m);
    stats[0] = m; stats[1] = z;
  }
  if (t == 1) {
    float m = -3.4e38f; for (int i = 0; i < n; ++i) m = fmaxf(m, p2[i]);
    float z = 0.f;      for (int i = 0; i < n; ++i) z += __expf(p2[i] - m);
    stats[2] = m; stats[3] = z;
  }
  __syncthreads();
  if (t < n) {
    p1[t] = __expf(p1[t] - stats[0]) / stats[1];
    p2[t] = __expf(p2[t] - stats[2]) / stats[3];
  }
  __syncthreads();
  float a1 = 0.f, a2 = 0.f;
  for (int i = 0; i < n; ++i) {
    float xv = x[(size_t)(s0 + i) * D + t];
    a1 += p1[i] * xv; a2 += p2[i] * xv;
  }
  out1[(size_t)b * D + t] = a1;
  out2[(size_t)b * D + t] = a2;
}

// ---------------------------------------------------------------------------
__device__ inline float blockReduce(float v, int op) {  // op: 0 sum, 1 max
  __shared__ float tmp[8];
  for (int o = 16; o > 0; o >>= 1) {
    float u = __shfl_down(v, o, 32);
    v = op ? fmaxf(v, u) : v + u;
  }
  int lane = threadIdx.x & 31, w = threadIdx.x >> 5, nw = blockDim.x >> 5;
  __syncthreads();
  if (lane == 0) tmp[w] = v;
  __syncthreads();
  if (threadIdx.x == 0) {
    float r = tmp[0];
    for (int i = 1; i < nw; ++i) r = op ? fmaxf(r, tmp[i]) : r + tmp[i];
    tmp[0] = r;
  }
  __syncthreads();
  float res = tmp[0];
  __syncthreads();
  return res;
}

__device__ inline float logsigf(float x) {
  return fminf(x, 0.f) - log1pf(__expf(-fabsf(x)));
}

// p_b = dot(z_d[b], u_d) + b_d3; acc += expand*logsig(p) + (1-expand)*logsig(1-p)
__global__ __launch_bounds__(128) void topo_loss(
    const float* __restrict__ z_d, const float* __restrict__ u_d,
    const float* __restrict__ b_d3, const float* __restrict__ expand,
    float* __restrict__ acc)
{
  int b = blockIdx.x, t = threadIdx.x;
  float v = z_d[(size_t)b * D + t] * u_d[t];
  float p = blockReduce(v, 0);
  if (t == 0) {
    p += b_d3[0];
    float ex = expand[b];
    atomicAdd(acc, ex * logsigf(p) + (1.f - ex) * logsigf(1.f - p));
  }
}

// acc += log_softmax(q[b])[wid[b]]
__global__ __launch_bounds__(256) void label_loss(
    const float* __restrict__ q, const int* __restrict__ wid,
    float* __restrict__ acc, int V)
{
  int b = blockIdx.x, t = threadIdx.x;
  const float* row = q + (size_t)b * V;
  float m = -3.4e38f;
  for (int k = t; k < V; k += 256) m = fmaxf(m, row[k]);
  m = blockReduce(m, 1);
  float s = 0.f;
  for (int k = t; k < V; k += 256) s += __expf(row[k] - m);
  s = blockReduce(s, 0);
  if (t == 0) atomicAdd(acc, row[wid[b]] - m - logf(s));
}

__global__ void finalize(const float* __restrict__ accs, float* __restrict__ out, float invB) {
  out[0] = -accs[0] * invB;
  out[1] = -accs[1] * invB;
}

// ---------------------------------------------------------------------------
extern "C" void kernel_launch(void* const* d_in, const int* in_sizes, int n_in,
                              void* d_out, int out_size, void* d_ws, size_t ws_size,
                              hipStream_t stream)
{
  const float* msg    = (const float*)d_in[0];
  const float* f_src  = (const float*)d_in[1];
  const float* f_dst  = (const float*)d_in[2];
  const float* x_T    = (const float*)d_in[3];
  const float* x_G    = (const float*)d_in[4];
  const float* expand = (const float*)d_in[5];
  const float* wz  = (const float*)d_in[6];
  const float* uz  = (const float*)d_in[7];
  const float* bz  = (const float*)d_in[8];
  const float* wr  = (const float*)d_in[9];
  const float* ur  = (const float*)d_in[10];
  const float* br  = (const float*)d_in[11];
  const float* w   = (const float*)d_in[12];
  const float* u   = (const float*)d_in[13];
  const float* bb  = (const float*)d_in[14];
  const float* w_d1= (const float*)d_in[15];
  const float* w_d2= (const float*)d_in[16];
  const float* b_d1= (const float*)d_in[17];
  const float* a_dT= (const float*)d_in[18];
  const float* a_dG= (const float*)d_in[19];
  const float* w_d3= (const float*)d_in[20];
  const float* w_d4= (const float*)d_in[21];
  const float* b_d2= (const float*)d_in[22];
  const float* u_d = (const float*)d_in[23];
  const float* b_d3= (const float*)d_in[24];
  const float* w_l1= (const float*)d_in[25];
  const float* w_l2= (const float*)d_in[26];
  const float* b_l1= (const float*)d_in[27];
  const float* a_lT= (const float*)d_in[28];
  const float* a_lG= (const float*)d_in[29];
  const float* u_l = (const float*)d_in[30];
  const float* b_l2= (const float*)d_in[31];
  const int* lg_src = (const int*)d_in[32];
  const int* lg_dst = (const int*)d_in[33];
  const int* eids   = (const int*)d_in[34];
  const int* wid    = (const int*)d_in[35];
  const int* bnnT   = (const int*)d_in[36];
  const int* bnnG   = (const int*)d_in[37];

  const int E   = in_sizes[0] / D;
  const int ELG = in_sizes[32];
  const int B   = in_sizes[34];
  const int V   = in_sizes[30] / D;

  const size_t dE = (size_t)E * D;
  const size_t BD = (size_t)B * D;

  // workspace: 3 big E*128 buffers; small area carved out of z after the GRU.
  float* s_buf   = (float*)d_ws;     // s, later aliased by msg_new
  float* z_buf   = s_buf + dE;       // z, later the small-intermediate area
  float* srh_buf = z_buf + dE;       // srh, later sum_h

  float* SM   = z_buf;
  float* h_t  = SM + 0 * BD;
  float* hpTd = SM + 1 * BD;
  float* hpGd = SM + 2 * BD;
  float* hpTl = SM + 3 * BD;
  float* hpGl = SM + 4 * BD;
  float* cTd  = SM + 5 * BD;
  float* cGd  = SM + 6 * BD;
  float* cTl  = SM + 7 * BD;
  float* cGl  = SM + 8 * BD;
  float* z_d  = SM + 9 * BD;
  float* z_l  = SM + 10 * BD;
  float* q    = SM + 11 * BD;            // B*V floats
  float* accs = q + (size_t)B * V;       // 2 floats
  int*   offsT = (int*)(accs + 2);       // B+1 ints
  int*   offsG = offsT + (B + 1);        // B+1 ints

  const int rbE   = (E + 255) / 256;
  const int rbELG = (ELG + 255) / 256;
  const int rbB   = (B + 255) / 256;
  const int scatBlocks = (int)(((size_t)ELG * 32 + 255) / 256);

  const float* NF = nullptr; const int* NI = nullptr; float* NFW = nullptr;

  // 1) s = segment_sum(msg[lg_src], lg_dst)
  hipMemsetAsync(s_buf, 0, dE * sizeof(float), stream);
  scatter_sum<<<scatBlocks, 256, 0, stream>>>(msg, lg_src, lg_dst, s_buf, ELG);

  // 2) z = sigmoid(f_src@wz + s@uz + bz)
  wmma_gemm<<<dim3(rbE, 1), 256, 0, stream>>>(E, D, D, D,
      f_src, NI, wz, s_buf, NI, uz, NF, NI, NF,
      bz, 1, 0, z_buf, NF, NI, NI, NFW, NF, NF);

  // 3) srh = segment_sum(sigmoid(f_dst[lg_src]@wr + msg[lg_src]@ur + br) * msg[lg_src], lg_dst)
  hipMemsetAsync(srh_buf, 0, dE * sizeof(float), stream);
  wmma_gemm<<<dim3(rbELG, 1), 256, 0, stream>>>(ELG, D, D, D,
      f_dst, lg_src, wr, msg, lg_src, ur, NF, NI, NF,
      br, 1, 1, NFW, msg, lg_src, lg_dst, srh_buf, NF, NF);

  // 4) msg_new = (1-z)*s + z*tanh(f_src@w + srh@u + b)   (written over s in place)
  wmma_gemm<<<dim3(rbE, 1), 256, 0, stream>>>(E, D, D, D,
      f_src, NI, w, srh_buf, NI, u, NF, NI, NF,
      bb, 0, 2, s_buf, NF, NI, NI, NFW, z_buf, s_buf);

  // 5) sum_h = segment_sum(msg_new[lg_src], lg_dst)   (into srh region, now free)
  hipMemsetAsync(srh_buf, 0, dE * sizeof(float), stream);
  scatter_sum<<<scatBlocks, 256, 0, stream>>>(s_buf, lg_src, lg_dst, srh_buf, ELG);

  // node offsets for segment softmax
  scan_offsets<<<1, 256, 0, stream>>>(bnnT, offsT, B);
  scan_offsets<<<1, 256, 0, stream>>>(bnnG, offsG, B);
  hipMemsetAsync(accs, 0, 2 * sizeof(float), stream);

  // 6) h_t = relu(f_src[eids]@w_d1 + sum_h[eids]@w_d2 + b_d1)
  wmma_gemm<<<dim3(rbB, 1), 256, 0, stream>>>(B, D, D, D,
      f_src, eids, w_d1, srh_buf, eids, w_d2, NF, NI, NF,
      b_d1, 2, 0, h_t, NF, NI, NI, NFW, NF, NF);

  // 7) attention projections (h@a): topology from h_t, label from m_e = msg_new[eids]
  wmma_gemm<<<dim3(rbB, 1), 256, 0, stream>>>(B, D, D, D,
      h_t, NI, a_dT, NF, NI, NF, NF, NI, NF, NF, 0, 0, hpTd, NF, NI, NI, NFW, NF, NF);
  wmma_gemm<<<dim3(rbB, 1), 256, 0, stream>>>(B, D, D, D,
      h_t, NI, a_dG, NF, NI, NF, NF, NI, NF, NF, 0, 0, hpGd, NF, NI, NI, NFW, NF, NF);
  wmma_gemm<<<dim3(rbB, 1), 256, 0, stream>>>(B, D, D, D,
      s_buf, eids, a_lT, NF, NI, NF, NF, NI, NF, NF, 0, 0, hpTl, NF, NI, NI, NFW, NF, NF);
  wmma_gemm<<<dim3(rbB, 1), 256, 0, stream>>>(B, D, D, D,
      s_buf, eids, a_lG, NF, NI, NF, NF, NI, NF, NF, 0, 0, hpGl, NF, NI, NI, NFW, NF, NF);

  // 8) dual attentions over x_T and x_G (topology + label fused per x pass)
  attn_dual<<<B, 128, 0, stream>>>(x_T, offsT, hpTd, hpTl, cTd, cTl);
  attn_dual<<<B, 128, 0, stream>>>(x_G, offsG, hpGd, hpGl, cGd, cGl);

  // 9) z_d = relu(h_t@w_d3 + cTd@w_d4[:128] + cGd@w_d4[128:] + b_d2)
  wmma_gemm<<<dim3(rbB, 1), 256, 0, stream>>>(B, D, D, D,
      h_t, NI, w_d3, cTd, NI, w_d4, cGd, NI, w_d4 + D * D,
      b_d2, 2, 0, z_d, NF, NI, NI, NFW, NF, NF);

  // 10) topology BCE-style loss
  topo_loss<<<B, 128, 0, stream>>>(z_d, u_d, b_d3, expand, accs);

  // 11) z_l = relu(m_e@w_l1 + cTl@w_l2[:128] + cGl@w_l2[128:] + b_l1)
  wmma_gemm<<<dim3(rbB, 1), 256, 0, stream>>>(B, D, D, D,
      s_buf, eids, w_l1, cTl, NI, w_l2, cGl, NI, w_l2 + D * D,
      b_l1, 2, 0, z_l, NF, NI, NI, NFW, NF, NF);

  // 12) q = z_l @ u_l + b_l2   (N = V = 800, column-chunked)
  wmma_gemm<<<dim3(rbB, (V + 127) / 128), 256, 0, stream>>>(B, V, V, V,
      z_l, NI, u_l, NF, NI, NF, NF, NI, NF,
      b_l2, 0, 0, q, NF, NI, NI, NFW, NF, NF);

  // 13) label cross-entropy
  label_loss<<<B, 256, 0, stream>>>(q, wid, accs + 1, V);

  // 14) write both losses
  finalize<<<1, 1, 0, stream>>>(accs, (float*)d_out, 1.0f / (float)B);
}
